// SoftDecisionTree_24051816858149
// MI455X (gfx1250) — matrix operand order repaired and checked by reading, not compiled, not run on hardware
//
#include <hip/hip_runtime.h>
#include <hip/hip_bf16.h>

typedef __attribute__((ext_vector_type(16))) _Float16 v16h;
typedef __attribute__((ext_vector_type(8)))  float    v8f;
typedef __attribute__((ext_vector_type(4)))  float    f4;

#define TREE_DEPTH 10
#define NODES      1023           // internal nodes
#define LEAVES     1024
#define DIMK       64
#define ROWS       32             // rows (batch elems) per block
#define GSTRIDE    1028           // padded LDS gate row stride (floats)
// sigmoid((logit)/T) = rcp(1 + exp2(logit * (-log2(e)/T)))
#define NEG_LOG2E_OVER_T (-1.4426950408889634f * 1.4285714285714286f)

__global__ __launch_bounds__(512)
void sdt_fused_kernel(const float* __restrict__ x,
                      const float* __restrict__ dec,
                      const float* __restrict__ bias,
                      float* __restrict__ out)
{
    extern __shared__ float gates[];          // [ROWS][GSTRIDE]
    const int tid  = threadIdx.x;
    const int lane = tid & 31;
    const int wave = tid >> 5;                // 0..15
    const int R0   = blockIdx.x * ROWS;

    // ---------------- Phase 1: gate logits GEMM via WMMA + sigmoid ----------------
    {
        const int rg    = wave >> 3;          // row group 0..1 (16 rows each)
        const int tbase = (wave & 7) * 8;     // this wave's 8 node tiles
        const int mrow  = lane & 15;          // M / N index within tile
        const int hi    = lane >> 4;          // half-wave select

        const float* xr  = x + (size_t)(R0 + rg * 16 + mrow) * DIMK;
        const f4*    xr4 = (const f4*)xr;

        // A fragments (16-bit A 16x32 layout): lane holds K = kb+{0..7} in v[0..7],
        // K = kb+16+{0..7} in v[8..15], kb = hi*8. Second fragment: +32.
        // All runs are 16B-aligned -> b128 loads.
        f4 ar[8];
        {
            const int b = hi * 2;             // f4 index of kb
            ar[0] = xr4[b + 0];  ar[1] = xr4[b + 1];    // kb .. kb+7
            ar[2] = xr4[b + 4];  ar[3] = xr4[b + 5];    // kb+16 .. kb+23
            ar[4] = xr4[b + 8];  ar[5] = xr4[b + 9];    // kb+32 .. kb+39
            ar[6] = xr4[b + 12]; ar[7] = xr4[b + 13];   // kb+48 .. kb+55
        }
        v16h aLo, aHi;
        #pragma unroll
        for (int i = 0; i < 8; ++i) {
            aLo[i]     = (_Float16)ar[0 + i / 4][i & 3];
            aLo[i + 8] = (_Float16)ar[2 + i / 4][i & 3];
            aHi[i]     = (_Float16)ar[4 + i / 4][i & 3];
            aHi[i + 8] = (_Float16)ar[6 + i / 4][i & 3];
        }

        for (int tt = 0; tt < 8; ++tt) {
            const int t  = tbase + tt;
            const int n  = t * 16 + mrow;                  // node (column) this lane holds
            const int nn = (n < NODES) ? n : (NODES - 1);  // clamp pad column
            const f4* dr4 = (const f4*)(dec + (size_t)nn * DIMK);

            // B fragments (16-bit B 32x16): same K order as A: K = kb+{0..7}, kb+16+{0..7}, kb = hi*8
            f4 br[8];
            {
                const int b = hi * 2;         // f4 index of kb = hi*8 floats
                #pragma unroll
                for (int i = 0; i < 4; ++i) {
                    br[i]     = dr4[b + (i & 1) + (i >> 1) * 4];       // K step 0: kb..kb+7, kb+16..kb+23
                    br[4 + i] = dr4[8 + b + (i & 1) + (i >> 1) * 4];   // K step 1: 32+ same
                }
            }
            v16h bLo, bHi;
            #pragma unroll
            for (int i = 0; i < 16; ++i) {
                bLo[i] = (_Float16)br[i / 4][i & 3];
                bHi[i] = (_Float16)br[4 + i / 4][i & 3];
            }

            v8f c = {};
            c = __builtin_amdgcn_wmma_f32_16x16x32_f16(false, aLo, false, bLo,
                                                       (short)0, c, false, false);
            c = __builtin_amdgcn_wmma_f32_16x16x32_f16(false, aHi, false, bHi,
                                                       (short)0, c, false, false);

            // Fold bias + 1/T + log2e into one FMA feeding v_exp_f32 (exp2)
            const float zc = bias[nn] * NEG_LOG2E_OVER_T;
            // C/D layout: VGPR j -> row M = j + 8*hi, col N = lane%16
            #pragma unroll
            for (int j = 0; j < 8; ++j) {
                const int m = rg * 16 + j + hi * 8;       // local row 0..31
                const float e = __builtin_amdgcn_exp2f(
                                    __builtin_fmaf(c[j], NEG_LOG2E_OVER_T, zc));
                const float g = __builtin_amdgcn_rcpf(1.0f + e);
                gates[m * GSTRIDE + n] = g;               // n==1023 pad slot never read
            }
        }
    }
    __syncthreads();

    // ---------------- Phase 2: tree path products, normalize, stream out ----------------
    {
        const int r    = tid >> 4;          // local row 0..31
        const int cch  = tid & 15;          // leaf chunk index
        const int base = cch * 64;          // first leaf of chunk
        const float* gr = &gates[r * GSTRIDE];

        // Levels 0..3 are constant across a 64-aligned chunk
        float p = 1.0f;
        #pragma unroll
        for (int l = 0; l < 4; ++l) {
            const float g = gr[(1 << l) - 1 + (base >> (TREE_DEPTH - l))];
            p *= ((base >> (9 - l)) & 1) ? g : (1.0f - g);
        }

        float P[9];
        P[3] = p;
        float leaf[64];
        float s = 0.0f;

        // Walk leaf pairs; recompute only levels whose leaf-index bit flipped.
        #pragma unroll
        for (int m = 0; m < 32; ++m) {
            const int i = base + 2 * m;
            const int start = (m == 0) ? 4 : (8 - __builtin_ctz(m)); // in [4,8]
            float q = P[start - 1];
            #pragma unroll 8
            for (int l = start; l <= 8; ++l) {
                const float g = gr[(1 << l) - 1 + (i >> (TREE_DEPTH - l))];
                q *= ((i >> (9 - l)) & 1) ? g : (1.0f - g);
                P[l] = q;
            }
            const float g9 = gr[511 + (i >> 1)];   // level-9 gate shared by the pair
            const float e0 = q * (1.0f - g9);
            const float e1 = q * g9;
            leaf[2 * m]     = e0;
            leaf[2 * m + 1] = e1;
            s += e0 + e1;
        }

        // Row sum: reduce across the 16 lanes sharing this row (lanes 0-15 / 16-31)
        s += __shfl_xor(s, 1, 32);
        s += __shfl_xor(s, 2, 32);
        s += __shfl_xor(s, 4, 32);
        s += __shfl_xor(s, 8, 32);
        const float inv = __builtin_amdgcn_rcpf(s + 1e-9f);

        float* orow = out + (size_t)(R0 + r) * LEAVES + base;
        #pragma unroll
        for (int qi = 0; qi < 16; ++qi) {
            f4 v;
            v.x = leaf[4 * qi + 0] * inv;
            v.y = leaf[4 * qi + 1] * inv;
            v.z = leaf[4 * qi + 2] * inv;
            v.w = leaf[4 * qi + 3] * inv;
            // streaming write-once output: non-temporal to keep L2 for `dec`
            __builtin_nontemporal_store(v, (f4*)(orow + 4 * qi));
        }
    }
}

extern "C" void kernel_launch(void* const* d_in, const int* in_sizes, int n_in,
                              void* d_out, int out_size, void* d_ws, size_t ws_size,
                              hipStream_t stream) {
    (void)n_in; (void)d_ws; (void)ws_size; (void)out_size;
    const float* x    = (const float*)d_in[0];
    const float* dec  = (const float*)d_in[1];
    const float* bias = (const float*)d_in[2];
    float* out        = (float*)d_out;

    const int batch = in_sizes[0] / DIMK;            // 131072
    const size_t lds_bytes = (size_t)ROWS * GSTRIDE * sizeof(float); // ~128.5 KB

    // Opt in to >64KB dynamic LDS (harmless if already allowed)
    (void)hipFuncSetAttribute((const void*)sdt_fused_kernel,
                              hipFuncAttributeMaxDynamicSharedMemorySize,
                              (int)lds_bytes);

    dim3 grid(batch / ROWS);
    dim3 block(512);
    hipLaunchKernelGGL(sdt_fused_kernel, grid, block, lds_bytes, stream,
                       x, dec, bias, out);
}
